// SimpleTransformerWithMemory_11596411699468
// MI455X (gfx1250) — matrix operand order repaired and checked by reading, not compiled, not run on hardware
//
#include <hip/hip_runtime.h>
#include <hip/hip_bf16.h>

// ---------------- problem constants ----------------
#define VV    32000
#define DD    256
#define HH    8
#define LL    2
#define FFD   1024
#define SS    512
#define BB    32
#define MM    4096
#define NCC   10
#define KTOP  4
#define BSTOT (BB*SS)      // 16384
#define DHH   32           // head dim

typedef _Float16 h16;
typedef __attribute__((ext_vector_type(16))) _Float16 v16h;
typedef __attribute__((ext_vector_type(8)))  _Float16 v8h;
typedef __attribute__((ext_vector_type(8)))  float    v8f;

// ---------------- WMMA tile loaders (CDNA5 documented layouts) ----------------
// A 16x32 f16 tile (M x K). Lane l: row = l&15, group g = l>>4.
// g=0: halfs 0..7 -> K=0..7,  halfs 8..15 -> K=16..23
// g=1: halfs 0..7 -> K=8..15, halfs 8..15 -> K=24..31
__device__ __forceinline__ v16h ld_tileA(const h16* __restrict__ a00, int lda) {
  int lane = threadIdx.x & 31;
  int r = lane & 15, g = lane >> 4;
  const h16* p = a00 + (long long)r * lda + g * 8;
  v8h lo = *(const v8h*)(p);
  v8h hi = *(const v8h*)(p + 16);
  v16h o;
#pragma unroll
  for (int i = 0; i < 8; ++i) { o[i] = lo[i]; o[i + 8] = hi[i]; }
  return o;
}

// B 32x16 f16 tile (K x N) read from the LDS-staged panel sB[32n][32k].
// Lane l holds column N = n0 + (l&15); group g = l>>4 holds K = g*8..g*8+7 and 16+g*8..16+g*8+7.
__device__ __forceinline__ v16h ld_tileB_lds(const h16* sB, int n0) {
  int lane = threadIdx.x & 31;
  int r = lane & 15, g = lane >> 4;
  const h16* p = sB + (n0 + r) * 32 + g * 8;
  v8h lo = *(const v8h*)(p);
  v8h hi = *(const v8h*)(p + 16);
  v16h o;
#pragma unroll
  for (int i = 0; i < 8; ++i) { o[i] = lo[i]; o[i + 8] = hi[i]; }
  return o;
}

// ---------------- batched WMMA GEMM:  C = alpha*(A @ B^T) + bias, opt relu ----
// A: [M,K] f16 (lda, batch stride sA in halfs)
// B: [N,K] f16 (ldb, batch stride sB in halfs)   N % 32 == 0, K % 32 == 0
// C: [M,N] f32 (ldc, batch stride sC in floats)
// Block = 8 waves, macro-tile 128M x 32N. B panel (32N x 32K, 2KB) staged to LDS
// per K-step via async global->LDS copies (one b64 per thread), shared by all waves.
__global__ void k_gemm(const h16* __restrict__ A, int lda, long long sA,
                       const h16* __restrict__ Bm, int ldb, long long sB,
                       float* __restrict__ C, int ldc, long long sC,
                       int Mm, int Nn, int Kk, float alpha,
                       const float* __restrict__ bias, int relu)
{
  __shared__ __align__(16) h16 sBp[32 * 32];
  int ngroups = Nn >> 5;
  int mg = blockIdx.x / ngroups;
  int ng = blockIdx.x - mg * ngroups;
  int w = threadIdx.y;                   // wave 0..7
  int lane = threadIdx.x;                // 0..31
  int t = w * 32 + lane;                 // 0..255
  long long bt = blockIdx.y;

  int tm = mg * 8 + w;
  bool valid = (tm * 16) < Mm;           // wave-uniform

  const h16* Ab = A  + bt * sA + (long long)(tm * 16) * lda;
  const h16* Bb = Bm + bt * sB + (long long)(ng * 32) * ldb;

  // staging: thread t copies 8 bytes: B row (ng*32 + sn), halfs [k0+sk, k0+sk+4)
  int sn = t >> 3;
  int sk = (t & 7) * 4;
  unsigned ldsOff = (unsigned)(uintptr_t)(&sBp[sn * 32 + sk]);
  const h16* gRow = Bb + (long long)sn * ldb + sk;

  v8f acc0 = {0.f, 0.f, 0.f, 0.f, 0.f, 0.f, 0.f, 0.f};
  v8f acc1 = {0.f, 0.f, 0.f, 0.f, 0.f, 0.f, 0.f, 0.f};

  for (int k0 = 0; k0 < Kk; k0 += 32) {
    // async copy of the 32x32 B panel into LDS (ASYNCcnt path)
    unsigned long long ga = (unsigned long long)(uintptr_t)(gRow + k0);
    asm volatile("global_load_async_to_lds_b64 %0, %1, off"
                 :: "v"(ldsOff), "v"(ga) : "memory");
    asm volatile("s_wait_asynccnt 0x0" ::: "memory");
    __syncthreads();

    if (valid) {
      if (k0 + 32 < Kk) __builtin_prefetch(Ab + k0 + 32, 0, 1);
      v16h a  = ld_tileA(Ab + k0, lda);
      v16h b0 = ld_tileB_lds(sBp, 0);
      v16h b1 = ld_tileB_lds(sBp, 16);
      acc0 = __builtin_amdgcn_wmma_f32_16x16x32_f16(false, a, false, b0,
                                                    (short)0, acc0, false, false);
      acc1 = __builtin_amdgcn_wmma_f32_16x16x32_f16(false, a, false, b1,
                                                    (short)0, acc1, false, false);
    }
    __syncthreads();                      // protect sBp before next stage
  }

  if (valid) {
    int col = lane & 15, g = lane >> 4;
    // C/D layout: acc[r] -> row (tm*16 + r + 8*g)
#pragma unroll
    for (int j = 0; j < 2; ++j) {
      int ncol = ng * 32 + j * 16 + col;
      float bv = bias ? bias[ncol] : 0.f;
      float* Cb = C + bt * sC + (long long)(tm * 16 + g * 8) * ldc + ncol;
      v8f acc = j ? acc1 : acc0;
#pragma unroll
      for (int r = 0; r < 8; ++r) {
        float v = acc[r] * alpha + bv;
        if (relu) v = fmaxf(v, 0.f);
        Cb[(long long)r * ldc] = v;
      }
    }
  }
}

// ---------------- elementwise / reduction kernels ----------------
__global__ void k_f2h(const float* __restrict__ s, h16* __restrict__ d, long long n) {
  long long i = (long long)blockIdx.x * blockDim.x + threadIdx.x;
  long long st = (long long)gridDim.x * blockDim.x;
  for (; i < n; i += st) d[i] = (h16)s[i];
}

__global__ void k_zerof(float* __restrict__ p, long long n) {
  long long i = (long long)blockIdx.x * blockDim.x + threadIdx.x;
  long long st = (long long)gridDim.x * blockDim.x;
  for (; i < n; i += st) p[i] = 0.f;
}

__global__ void k_embed(const int* __restrict__ ids, const float* __restrict__ tok,
                        const float* __restrict__ pos, float* __restrict__ x,
                        h16* __restrict__ xh)
{
  long long i = (long long)blockIdx.x * 256 + threadIdx.x;   // over B*S*D
  int d = (int)(i & 255);
  long long bs = i >> 8;
  int s = (int)(bs & (SS - 1));
  int id = ids[bs];
  float v = tok[(long long)id * DD + d] + pos[(long long)s * DD + d];
  x[i] = v; xh[i] = (h16)v;
}

__global__ void k_split(const float* __restrict__ qkv, h16* __restrict__ Qh,
                        h16* __restrict__ Kh, h16* __restrict__ VT)
{
  long long i = (long long)blockIdx.x * 256 + threadIdx.x;   // over B*S*D
  int d = (int)(i & 255);
  long long bs = i >> 8;
  int s = (int)(bs & (SS - 1));
  int b = (int)(bs >> 9);
  int h = d >> 5, dd = d & 31;
  const float* q = qkv + bs * (3 * DD);
  long long bh = (long long)(b * HH + h);
  Qh[(bh * SS + s) * DHH + dd] = (h16)q[d];
  Kh[(bh * SS + s) * DHH + dd] = (h16)q[DD + d];
  VT[(bh * DHH + dd) * SS + s] = (h16)q[2 * DD + d];
}

// softmax over a 512-row, writes f16 result in-place at the row start
__global__ void k_softmax(float* __restrict__ sc) {
  __shared__ float red[128];
  long long row = blockIdx.x;                 // H*S rows
  float* p = sc + row * SS;
  int t = threadIdx.x;                        // 128 threads
  float vals[4], e[4];
  float mx = -3.4e38f;
#pragma unroll
  for (int j = 0; j < 4; ++j) { vals[j] = p[t + j * 128]; mx = fmaxf(mx, vals[j]); }
  red[t] = mx; __syncthreads();
  for (int s = 64; s > 0; s >>= 1) { if (t < s) red[t] = fmaxf(red[t], red[t + s]); __syncthreads(); }
  mx = red[0]; __syncthreads();
  float sum = 0.f;
#pragma unroll
  for (int j = 0; j < 4; ++j) { e[j] = expf(vals[j] - mx); sum += e[j]; }
  red[t] = sum; __syncthreads();
  for (int s = 64; s > 0; s >>= 1) { if (t < s) red[t] += red[t + s]; __syncthreads(); }
  float inv = 1.f / red[0];
  __syncthreads();                            // all reads of the f32 row are done
  h16* ph = (h16*)p;
#pragma unroll
  for (int j = 0; j < 4; ++j) ph[t + j * 128] = (h16)(e[j] * inv);
}

// out = LayerNorm(x + o) * g + b ; writes f32 (in-place safe) and f16 copies
__global__ void k_add_ln(const float* __restrict__ x, const float* __restrict__ o,
                         const float* __restrict__ g, const float* __restrict__ bta,
                         float* __restrict__ outx, h16* __restrict__ outh)
{
  __shared__ float red[256];
  long long row = blockIdx.x;
  int t = threadIdx.x;
  float v = x[row * DD + t] + o[row * DD + t];
  red[t] = v; __syncthreads();
  for (int s = 128; s > 0; s >>= 1) { if (t < s) red[t] += red[t + s]; __syncthreads(); }
  float mean = red[0] * (1.f / DD); __syncthreads();
  float d = v - mean;
  red[t] = d * d; __syncthreads();
  for (int s = 128; s > 0; s >>= 1) { if (t < s) red[t] += red[t + s]; __syncthreads(); }
  float inv = rsqrtf(red[0] * (1.f / DD) + 1e-5f);
  float r = d * inv * g[t] + bta[t];
  outx[row * DD + t] = r; outh[row * DD + t] = (h16)r;
}

// probs = sigmoid(g1 . Wg2 + bg2), one wave per row
__global__ void k_gate2(const float* __restrict__ g1, const float* __restrict__ Wg2,
                        const float* __restrict__ bg2, float* __restrict__ probs)
{
  int row = blockIdx.x * 8 + (threadIdx.x >> 5);
  int lane = threadIdx.x & 31;
  const float* p = g1 + (long long)row * DD;
  float s = 0.f;
  for (int d = lane; d < DD; d += 32) s += p[d] * Wg2[d];
  for (int off = 16; off; off >>= 1) s += __shfl_xor(s, off, 32);
  if (lane == 0) probs[row] = 1.f / (1.f + expf(-(s + bg2[0])));
}

// single-block inclusive scan over 16384 mask bits (1024 threads x 16 elems)
__global__ void k_scan(const float* __restrict__ probs, int* __restrict__ maskA,
                       int* __restrict__ slotA, int* __restrict__ countA)
{
  __shared__ int ss[1024];
  int t = threadIdx.x;
  int base = t * 16;
  int m[16]; int run = 0;
#pragma unroll
  for (int j = 0; j < 16; ++j) { m[j] = probs[base + j] > 0.5f ? 1 : 0; run += m[j]; }
  ss[t] = run; __syncthreads();
  for (int off = 1; off < 1024; off <<= 1) {
    int v = (t >= off) ? ss[t - off] : 0;
    __syncthreads();
    ss[t] += v;
    __syncthreads();
  }
  int c = ss[t] - run;   // exclusive prefix
#pragma unroll
  for (int j = 0; j < 16; ++j) {
    c += m[j];
    slotA[base + j] = c - 1;
    maskA[base + j] = m[j];
  }
  if (t == 1023) countA[0] = ss[1023];
}

__global__ void k_scatter(const float* __restrict__ x32, const int* __restrict__ maskA,
                          const int* __restrict__ slotA, float* __restrict__ mem)
{
  int i = blockIdx.x;
  if (!maskA[i]) return;
  int s = slotA[i];
  if (s >= MM) return;
  mem[(long long)s * DD + threadIdx.x] = x32[(long long)i * DD + threadIdx.x];
}

// row-normalize src (row stride rstride) into f16 dst[rows, 256]
__global__ void k_rownorm(const float* __restrict__ src, long long rstride,
                          h16* __restrict__ dst)
{
  __shared__ float red[256];
  long long r = blockIdx.x;
  int t = threadIdx.x;
  float v = src[r * rstride + t];
  red[t] = v * v; __syncthreads();
  for (int s = 128; s > 0; s >>= 1) { if (t < s) red[t] += red[t + s]; __syncthreads(); }
  float inv = 1.f / (sqrtf(red[0]) + 1e-8f);
  dst[r * DD + t] = (h16)(v * inv);
}

// per-row top-4 over sims, softmax, weighted gather from mem, add cls_state
__global__ void k_topk(const float* __restrict__ sims, const int* __restrict__ countA,
                       const float* __restrict__ mem, const float* __restrict__ x32,
                       float* __restrict__ cls_aug)
{
  __shared__ float sv[512];
  __shared__ int   si[512];
  __shared__ float swt[KTOP];
  __shared__ int   sid[KTOP];
  int b = blockIdx.x, t = threadIdx.x;    // 128 threads
  int cnt = countA[0]; if (cnt > MM) cnt = MM;
  float bv[KTOP]; int bi[KTOP];
#pragma unroll
  for (int k = 0; k < KTOP; ++k) { bv[k] = -3.4e38f; bi[k] = 0; }
  for (int j = t; j < MM; j += 128) {
    float s = (j < cnt) ? sims[(long long)b * MM + j] : -1e9f;
    if (s > bv[KTOP - 1]) {
      int k = KTOP - 1;
      while (k > 0 && s > bv[k - 1]) { bv[k] = bv[k - 1]; bi[k] = bi[k - 1]; --k; }
      bv[k] = s; bi[k] = j;
    }
  }
#pragma unroll
  for (int k = 0; k < KTOP; ++k) { sv[t * KTOP + k] = bv[k]; si[t * KTOP + k] = bi[k]; }
  __syncthreads();
  if (t == 0) {
    float gv[KTOP]; int gi[KTOP];
#pragma unroll
    for (int k = 0; k < KTOP; ++k) { gv[k] = -3.4e38f; gi[k] = 0; }
    for (int j = 0; j < 512; ++j) {
      float s = sv[j];
      if (s > gv[KTOP - 1]) {
        int k = KTOP - 1;
        while (k > 0 && s > gv[k - 1]) { gv[k] = gv[k - 1]; gi[k] = gi[k - 1]; --k; }
        gv[k] = s; gi[k] = si[j];
      }
    }
    float sum = 0.f, wv[KTOP];
#pragma unroll
    for (int k = 0; k < KTOP; ++k) { wv[k] = expf(gv[k] - gv[0]); sum += wv[k]; }
#pragma unroll
    for (int k = 0; k < KTOP; ++k) { swt[k] = wv[k] / sum; sid[k] = gi[k]; }
  }
  __syncthreads();
  for (int d = t; d < DD; d += 128) {
    float acc = 0.f;
#pragma unroll
    for (int k = 0; k < KTOP; ++k) acc += swt[k] * mem[(long long)sid[k] * DD + d];
    float cs = x32[(long long)b * SS * DD + d];   // cls_state = h[:,0,:]
    cls_aug[b * DD + d] = cs + (cnt > 0 ? acc : 0.f);
  }
}

__global__ void k_cls(const float* __restrict__ cls_aug, const float* __restrict__ Wc,
                      const float* __restrict__ bc, float* __restrict__ out)
{
  __shared__ float red[256];
  int b = blockIdx.x, t = threadIdx.x;
  float x = cls_aug[b * DD + t];
  for (int c = 0; c < NCC; ++c) {
    red[t] = x * Wc[c * DD + t];
    __syncthreads();
    for (int s = 128; s > 0; s >>= 1) { if (t < s) red[t] += red[t + s]; __syncthreads(); }
    if (t == 0) out[b * NCC + c] = red[0] + bc[c];
    __syncthreads();
  }
}

// ---------------- host-side helpers ----------------
static void launch_gemm(hipStream_t st, const h16* A, int lda, long long sA,
                        const h16* Bm, int ldb, long long sB,
                        float* C, int ldc, long long sC,
                        int M, int N, int K, float alpha,
                        const float* bias, int relu, int batches)
{
  int mgroups = (M + 127) / 128;
  int ngroups = N / 32;
  dim3 grid(mgroups * ngroups, batches);
  dim3 block(32, 8);
  k_gemm<<<grid, block, 0, st>>>(A, lda, sA, Bm, ldb, sB, C, ldc, sC,
                                 M, N, K, alpha, bias, relu);
}

extern "C" void kernel_launch(void* const* d_in, const int* in_sizes, int n_in,
                              void* d_out, int out_size, void* d_ws, size_t ws_size,
                              hipStream_t stream)
{
  const int*   ids   = (const int*)  d_in[0];
  const float* tok   = (const float*)d_in[1];
  const float* pos   = (const float*)d_in[2];
  const float* Wqkv  = (const float*)d_in[3];
  const float* bqkv  = (const float*)d_in[4];
  const float* Wo    = (const float*)d_in[5];
  const float* bo    = (const float*)d_in[6];
  const float* ln1g  = (const float*)d_in[7];
  const float* ln1b  = (const float*)d_in[8];
  const float* W1    = (const float*)d_in[9];
  const float* b1    = (const float*)d_in[10];
  const float* W2    = (const float*)d_in[11];
  const float* b2    = (const float*)d_in[12];
  const float* ln2g  = (const float*)d_in[13];
  const float* ln2b  = (const float*)d_in[14];
  const float* Wg1   = (const float*)d_in[15];
  const float* bg1   = (const float*)d_in[16];
  const float* Wg2   = (const float*)d_in[17];
  const float* bg2   = (const float*)d_in[18];
  const float* Wc    = (const float*)d_in[19];
  const float* bc    = (const float*)d_in[20];
  float* out = (float*)d_out;

  // bump allocator over workspace
  char* w = (char*)d_ws;
  auto bump = [&](size_t bytes) -> char* {
    char* r = w; w += (bytes + 255) & ~(size_t)255; return r;
  };
  float* x32   = (float*)bump((size_t)BSTOT * DD * 4);
  h16*   xh    = (h16*)  bump((size_t)BSTOT * DD * 2);
  float* qkv32 = (float*)bump((size_t)BSTOT * 3 * DD * 4);
  float* p32   = (float*)bump((size_t)BSTOT * DD * 4);
  float* o32   = (float*)bump((size_t)BSTOT * DD * 4);      // attn out / gate hidden
  h16*   oh    = (h16*)  bump((size_t)BSTOT * DD * 2);
  float* ff32  = (float*)bump((size_t)BSTOT * FFD * 4);
  h16*   ffh   = (h16*)  bump((size_t)BSTOT * FFD * 2);
  h16*   Qh    = (h16*)  bump((size_t)BB * HH * SS * DHH * 2);
  h16*   Kh    = (h16*)  bump((size_t)BB * HH * SS * DHH * 2);
  h16*   VT    = (h16*)  bump((size_t)BB * HH * DHH * SS * 2);
  float* sc    = (float*)bump((size_t)HH * SS * SS * 4);    // per-batch scores
  h16*   WqkvH = (h16*)  bump((size_t)LL * 3 * DD * DD * 2);
  h16*   WoH   = (h16*)  bump((size_t)LL * DD * DD * 2);
  h16*   W1H   = (h16*)  bump((size_t)LL * FFD * DD * 2);
  h16*   W2H   = (h16*)  bump((size_t)LL * DD * FFD * 2);
  h16*   Wg1H  = (h16*)  bump((size_t)DD * DD * 2);
  float* probs = (float*)bump((size_t)BSTOT * 4);
  int*   maskA = (int*)  bump((size_t)BSTOT * 4);
  int*   slotA = (int*)  bump((size_t)BSTOT * 4);
  int*   cntA  = (int*)  bump(256);
  float* memB  = (float*)bump((size_t)MM * DD * 4);
  h16*   knh   = (h16*)  bump((size_t)MM * DD * 2);
  h16*   qnh   = (h16*)  bump((size_t)BB * DD * 2);
  float* sims  = (float*)bump((size_t)BB * MM * 4);
  float* caug  = (float*)bump((size_t)BB * DD * 4);
  (void)ws_size; (void)in_sizes; (void)n_in; (void)out_size;

  // weights -> f16 (recomputed every call; deterministic)
  k_f2h<<<2048, 256, 0, stream>>>(Wqkv, WqkvH, (long long)LL * 3 * DD * DD);
  k_f2h<<<2048, 256, 0, stream>>>(Wo,   WoH,   (long long)LL * DD * DD);
  k_f2h<<<2048, 256, 0, stream>>>(W1,   W1H,   (long long)LL * FFD * DD);
  k_f2h<<<2048, 256, 0, stream>>>(W2,   W2H,   (long long)LL * DD * FFD);
  k_f2h<<<2048, 256, 0, stream>>>(Wg1,  Wg1H,  (long long)DD * DD);

  // embeddings
  k_embed<<<BSTOT, 256, 0, stream>>>(ids, tok, pos, x32, xh);

  const float scl = 0.17677669529663687f;  // 1/sqrt(32)

  for (int l = 0; l < LL; ++l) {
    // qkv = x @ Wqkv^T + bqkv
    launch_gemm(stream, xh, DD, 0, WqkvH + (long long)l * 3 * DD * DD, DD, 0,
                qkv32, 3 * DD, 0, BSTOT, 3 * DD, DD, 1.f, bqkv + l * 3 * DD, 0, 1);
    k_split<<<BSTOT, 256, 0, stream>>>(qkv32, Qh, Kh, VT);

    for (int b = 0; b < BB; ++b) {
      const h16* Qb = Qh + (long long)b * HH * SS * DHH;
      const h16* Kb = Kh + (long long)b * HH * SS * DHH;
      // scores[h] = Q @ K^T * scl      (batched over H heads)
      launch_gemm(stream, Qb, DHH, (long long)SS * DHH,
                  Kb, DHH, (long long)SS * DHH,
                  sc, SS, (long long)SS * SS,
                  SS, SS, DHH, scl, (const float*)nullptr, 0, HH);
      k_softmax<<<HH * SS, 128, 0, stream>>>(sc);   // f16 attn written in-place
      // o[b] = attn @ V   (A rows are f16 packed at stride 1024 halfs)
      launch_gemm(stream, (const h16*)sc, 2 * SS, (long long)SS * SS * 2,
                  VT + (long long)b * HH * DHH * SS, SS, (long long)DHH * SS,
                  o32 + (long long)b * SS * DD, DD, (long long)DHH,
                  SS, DHH, SS, 1.f, (const float*)nullptr, 0, HH);
    }

    // output projection + residual + LN1
    k_f2h<<<4096, 256, 0, stream>>>(o32, oh, (long long)BSTOT * DD);
    launch_gemm(stream, oh, DD, 0, WoH + (long long)l * DD * DD, DD, 0,
                p32, DD, 0, BSTOT, DD, DD, 1.f, bo + l * DD, 0, 1);
    k_add_ln<<<BSTOT, 256, 0, stream>>>(x32, p32, ln1g + l * DD, ln1b + l * DD, x32, xh);

    // FFN + residual + LN2
    launch_gemm(stream, xh, DD, 0, W1H + (long long)l * FFD * DD, DD, 0,
                ff32, FFD, 0, BSTOT, FFD, DD, 1.f, b1 + l * FFD, 1, 1);
    k_f2h<<<8192, 256, 0, stream>>>(ff32, ffh, (long long)BSTOT * FFD);
    launch_gemm(stream, ffh, FFD, 0, W2H + (long long)l * DD * FFD, FFD, 0,
                p32, DD, 0, BSTOT, DD, FFD, 1.f, b2 + l * DD, 0, 1);
    k_add_ln<<<BSTOT, 256, 0, stream>>>(x32, p32, ln2g + l * DD, ln2b + l * DD, x32, xh);
  }

  // gate MLP -> probs
  launch_gemm(stream, xh, DD, 0, Wg1H, DD, 0, o32, DD, 0,
              BSTOT, DD, DD, 1.f, bg1, 1, 1);
  k_gate2<<<BSTOT / 8, 256, 0, stream>>>(o32, Wg2, bg2, probs);

  // mask -> cumsum -> scatter into memory
  k_scan<<<1, 1024, 0, stream>>>(probs, maskA, slotA, cntA);
  k_zerof<<<2048, 256, 0, stream>>>(memB, (long long)MM * DD);
  k_scatter<<<BSTOT, 256, 0, stream>>>(x32, maskA, slotA, memB);

  // normalized query/key, sims = qn @ kn^T (WMMA)
  k_rownorm<<<MM, 256, 0, stream>>>(memB, DD, knh);
  k_rownorm<<<BB, 256, 0, stream>>>(x32, (long long)SS * DD, qnh);
  launch_gemm(stream, qnh, DD, 0, knh, DD, 0, sims, MM, 0,
              BB, MM, DD, 1.f, (const float*)nullptr, 0, 1);

  // top-K attention over memory + classifier
  k_topk<<<BB, 128, 0, stream>>>(sims, cntA, memB, x32, caug);
  k_cls<<<BB, 256, 0, stream>>>(caug, Wc, bc, out);
}
